// ANIInteraction_21955872817670
// MI455X (gfx1250) — matrix operand order repaired and checked by reading, not compiled, not run on hardware
//
#include <hip/hip_runtime.h>

#define N_ATOMS   50000
#define AEV_DIM   1008
#define KPAD1     1024
#define IDX_STRIDE 50176                 // ints per species index list
#define CNT_BYTES  256
#define IDX_BYTES  (7 * IDX_STRIDE * 4)  // 1,404,928
#define W_OFF      (CNT_BYTES + IDX_BYTES) // 1,405,184 (16B aligned)
#define WSTRIDE    344064                // bf16 elems per species (max needed 342,176)

// per-species hidden sizes, ELEMENT_ORDER = H, C, O, N, S, F, Cl
static __device__ const int cH1[7] = {256, 224, 192, 192, 160, 160, 160};
static __device__ const int cH2[7] = {192, 192, 160, 160, 128, 128, 128};
static __device__ const int cH3[7] = {160, 160, 128, 128,  96,  96,  96};

typedef __attribute__((ext_vector_type(16))) __bf16 v16bf;
typedef __attribute__((ext_vector_type(8)))  float  v8f;

union Frag { v16bf v; uint4 q[2]; unsigned short u[16]; };

struct PP { const float* p[56]; };   // 7 species x (W1,b1,W2,b2,W3,b3,W4,b4)

__device__ __forceinline__ unsigned short f2bf(float f) {
    unsigned u = __float_as_uint(f);
    unsigned r = (u + 0x7FFFu + ((u >> 16) & 1u)) >> 16;   // round-to-nearest-even
    return (unsigned short)r;
}
__device__ __forceinline__ float bf2f(unsigned short u) {
    return __uint_as_float(((unsigned)u) << 16);
}
__device__ __forceinline__ float celu01(float x) {
    return x > 0.0f ? x : 0.1f * (__expf(x * 10.0f) - 1.0f);
}

// ---------------- kernel 1: zero species counters ----------------
__global__ void ani_zero_kernel(int* cnt) {
    if (threadIdx.x < 8) cnt[threadIdx.x] = 0;
}

// ---------------- kernel 2: fp32 [K][N] -> bf16 transposed [N][ldK] ----------------
__global__ void ani_wconv_kernel(PP pp, unsigned short* __restrict__ wts) {
    int s = blockIdx.y, L = blockIdx.z;
    int H1 = cH1[s], H2 = cH2[s], H3 = cH3[s];
    int N, ldK, srcK; size_t off;
    if (L == 0)      { N = H1; ldK = KPAD1; srcK = AEV_DIM; off = 0; }
    else if (L == 1) { N = H2; ldK = H1;    srcK = H1; off = (size_t)KPAD1 * H1; }
    else if (L == 2) { N = H3; ldK = H2;    srcK = H2; off = (size_t)KPAD1 * H1 + (size_t)H1 * H2; }
    else             { N = 1;  ldK = H3;    srcK = H3; off = (size_t)KPAD1 * H1 + (size_t)H1 * H2 + (size_t)H2 * H3; }
    const float* W = pp.p[s * 8 + 2 * L];
    unsigned short* dst = wts + (size_t)s * WSTRIDE + off;
    int total = N * ldK;
    for (int e = blockIdx.x * 256 + threadIdx.x; e < total; e += gridDim.x * 256) {
        int n = e / ldK, k = e - n * ldK;
        float v = (k < srcK) ? W[(size_t)k * N + n] : 0.0f;
        dst[e] = f2bf(v);
    }
}

// ---------------- kernel 3: bucket atoms by species ----------------
__global__ void ani_bucket_kernel(const int* __restrict__ species,
                                  int* cnt, int* __restrict__ idxAll, int n) {
    int i = blockIdx.x * 256 + threadIdx.x;
    if (i < n) {
        int s = species[i] & 7;
        int slot = atomicAdd(&cnt[s], 1);
        idxAll[s * IDX_STRIDE + slot] = i;
    }
}

// ---------------- WMMA GEMM layer: D(16xN) = A(16xK) * Wt^T + bias, CELU ----------------
__device__ __forceinline__ void gemm_layer(const unsigned short* sIn, int ldIn, int K,
                                           const unsigned short* __restrict__ W, int ldK,
                                           const float* __restrict__ bias, int N,
                                           unsigned short* sOut, int ldOut,
                                           int lane, int wave) {
    int nTiles = N >> 4;
    int m16 = lane & 15, hi = lane >> 4;
    const unsigned short* arow = sIn + m16 * ldIn + hi * 8;   // A: lane m, K halves {hi*8, 16+hi*8}
    for (int t = wave; t < nTiles; t += 8) {
        v8f acc = {};
        const unsigned short* wrow = W + (size_t)(t * 16 + m16) * ldK + hi * 8; // B: lane n, K halves {hi*8, 16+hi*8}
#pragma unroll 4
        for (int k0 = 0; k0 < K; k0 += 32) {
            Frag a, b;
            a.q[0] = *(const uint4*)(arow + k0);
            a.q[1] = *(const uint4*)(arow + k0 + 16);
            b.q[0] = *(const uint4*)(wrow + k0);
            b.q[1] = *(const uint4*)(wrow + k0 + 16);
            acc = __builtin_amdgcn_wmma_f32_16x16x32_bf16(false, a.v, false, b.v,
                                                          (short)0, acc, false, false);
        }
        int n = t * 16 + m16;
        float bv = bias[n];
#pragma unroll
        for (int v = 0; v < 8; v++) {
            float x = celu01(acc[v] + bv);
            sOut[(v + hi * 8) * ldOut + n] = f2bf(x);   // C/D layout: m = v + 8*hi, col = lane&15
        }
    }
}

// ---------------- kernel 4: fused per-species MLP over 16-atom tiles ----------------
__global__ __launch_bounds__(256)
void ani_mlp_kernel(const float* __restrict__ aev,
                    const int* __restrict__ cnt, const int* __restrict__ idxAll,
                    const unsigned short* __restrict__ wts, PP pp,
                    float* __restrict__ out) {
    __shared__ __align__(16) unsigned short sA[16 * KPAD1];  // 32 KB bf16 AEV tile (K-padded)
    __shared__ __align__(16) unsigned short sH0[16 * 256];   // 8 KB activation ping
    __shared__ __align__(16) unsigned short sH1[16 * 256];   // 8 KB activation pong
    __shared__ int sIdx[16];

    int s = blockIdx.y;
    int count = cnt[s];
    int base = blockIdx.x * 16;
    if (base >= count) return;

    int H1 = cH1[s], H2 = cH2[s], H3 = cH3[s];
    const int* idx = idxAll + s * IDX_STRIDE;
    int tid = threadIdx.x, lane = tid & 31, wave = tid >> 5;

    if (tid < 16) sIdx[tid] = (base + tid < count) ? idx[base + tid] : -1;
    __syncthreads();

    // gather 16 AEV rows -> bf16 in LDS, zero-pad K 1008..1023 and invalid rows
    for (int q = tid; q < 16 * 256; q += 256) {
        int row = q >> 8, k = (q & 255) * 4;
        int a = sIdx[row];
        uint2 packed;
        if (a >= 0 && k < AEV_DIM) {
            const float4 f = *(const float4*)(aev + (size_t)a * AEV_DIM + k);
            packed.x = (unsigned)f2bf(f.x) | ((unsigned)f2bf(f.y) << 16);
            packed.y = (unsigned)f2bf(f.z) | ((unsigned)f2bf(f.w) << 16);
        } else {
            packed.x = 0u; packed.y = 0u;
        }
        *(uint2*)(sA + row * KPAD1 + k) = packed;
    }
    __syncthreads();

    const unsigned short* wb = wts + (size_t)s * WSTRIDE;
    size_t off2 = (size_t)KPAD1 * H1;
    size_t off3 = off2 + (size_t)H1 * H2;
    size_t off4 = off3 + (size_t)H2 * H3;

    gemm_layer(sA,  KPAD1, KPAD1, wb,        KPAD1, pp.p[s * 8 + 1], H1, sH0, 256, lane, wave);
    __syncthreads();
    gemm_layer(sH0, 256,   H1,    wb + off2, H1,    pp.p[s * 8 + 3], H2, sH1, 256, lane, wave);
    __syncthreads();
    gemm_layer(sH1, 256,   H2,    wb + off3, H2,    pp.p[s * 8 + 5], H3, sH0, 256, lane, wave);
    __syncthreads();

    // final h3 -> scalar: tiny dot product per atom
    if (tid < 16) {
        int a = sIdx[tid];
        if (a >= 0) {
            const unsigned short* w4 = wb + off4;
            float sum = pp.p[s * 8 + 7][0];
            for (int k = 0; k < H3; k++)
                sum += bf2f(sH0[tid * 256 + k]) * bf2f(w4[k]);
            out[a] = sum;
        }
    }
}

extern "C" void kernel_launch(void* const* d_in, const int* in_sizes, int n_in,
                              void* d_out, int out_size, void* d_ws, size_t ws_size,
                              hipStream_t stream) {
    const int*   species = (const int*)d_in[0];
    const float* aev     = (const float*)d_in[1];
    PP pp;
    for (int i = 0; i < 56; i++) pp.p[i] = (const float*)d_in[2 + i];

    char* ws = (char*)d_ws;
    int* cnt = (int*)ws;
    int* idxAll = (int*)(ws + CNT_BYTES);
    unsigned short* wts = (unsigned short*)(ws + W_OFF);
    float* out = (float*)d_out;

    ani_zero_kernel<<<1, 256, 0, stream>>>(cnt);
    ani_wconv_kernel<<<dim3(1024, 7, 4), 256, 0, stream>>>(pp, wts);
    ani_bucket_kernel<<<(N_ATOMS + 255) / 256, 256, 0, stream>>>(species, cnt, idxAll, N_ATOMS);
    ani_mlp_kernel<<<dim3((N_ATOMS + 15) / 16, 7), 256, 0, stream>>>(aev, cnt, idxAll, wts, pp, out);
}
